// WideLSTM_77919296684777
// MI455X (gfx1250) — matrix operand order repaired and checked by reading, not compiled, not run on hardware
//
#include <hip/hip_runtime.h>
#include <hip/hip_bf16.h>

// ---------------------------------------------------------------------------
// Wide LSTM (N=32 blocks, I=64, H=128, B=16, T=256) for MI455X (gfx1250).
//
// Persistent-RNN design: one workgroup (16 wave32s) per LSTM block.
//  - Weights converted fp32->bf16 once, swizzled through LDS, then pinned in
//    VGPRs (12 x v16bf per lane) for all 256 timesteps.
//  - Per step: fused [x_t | h] (16x192) x [Wih|Whh]^T (192x512) GEMM via
//    v_wmma_f32_16x16x32_bf16 (f32 accumulate, bias folded into C init).
//    All 12 A-fragment ds_load_b128 are fenced ahead of the WMMA chain with
//    sched_barrier so they stay in distinct VGPRs and the WMMAs run behind
//    partial s_wait_dscnt thresholds instead of dscnt==0 stalls.
//  - f32 activations + state update, vectorized b128 LDS/HBM traffic.
//  - Two barriers per step; x_{t+1} staged + x_{t+2} prefetched each step.
// ---------------------------------------------------------------------------

#define LSTM_N   32
#define LSTM_I   64
#define LSTM_H   128
#define LSTM_B   16
#define LSTM_T   256
#define G4H      512      // 4*H
#define NTHREADS 512      // 16 wave32s

typedef __attribute__((ext_vector_type(16))) __bf16          v16bf;
typedef __attribute__((ext_vector_type(8)))  float           v8f;
typedef __attribute__((ext_vector_type(4)))  float           v4f;
typedef __attribute__((ext_vector_type(8)))  unsigned short  u16x8;
typedef __attribute__((ext_vector_type(16))) unsigned short  u16x16;
typedef __attribute__((ext_vector_type(4)))  unsigned short  u16x4;

// ----- LDS carve (bytes) ----------------------------------------------------
#define OFF_WIH   0                       // 32*2*32*16 bf16  = 65536 B
#define OFF_WHH   65536                   // 32*4*32*16 bf16  = 131072 B
#define OFF_BIAS  196608                  // 512 f32          = 2048 B
#define OFF_X     198656                  // 16*64 bf16       = 2048 B
#define OFF_HS    200704                  // 16*128 bf16      = 4096 B
#define OFF_CS    204800                  // 16*128 f32       = 8192 B
#define OFF_GATES 212992                  // 16*512 f32       = 32768 B
#define SMEM_BYTES 245760                 // 240 KB (<= 320 KB WGP LDS)

__device__ __forceinline__ unsigned short f2bf(float f) {
    unsigned int u = __builtin_bit_cast(unsigned int, f);
    u += 0x7FFFu + ((u >> 16) & 1u);      // round-to-nearest-even
    return (unsigned short)(u >> 16);
}

__device__ __forceinline__ v16bf ld_frag32(const unsigned short* p) {
    u16x8 lo = *(const u16x8*)(p);
    u16x8 hi = *(const u16x8*)(p + 8);
    u16x16 w = __builtin_shufflevector(lo, hi, 0,1,2,3,4,5,6,7,8,9,10,11,12,13,14,15);
    return __builtin_bit_cast(v16bf, w);
}

__device__ __forceinline__ float sigmoid_f(float x) {
    return __builtin_amdgcn_rcpf(1.0f + __expf(-x));
}
__device__ __forceinline__ float tanh_f(float x) {
    float e = __expf(2.0f * x);
    return (e - 1.0f) * __builtin_amdgcn_rcpf(e + 1.0f);
}

__global__ void __launch_bounds__(NTHREADS, 1)
wide_lstm_kernel(const float* __restrict__ x,      // [B,T,N*I]
                 const float* __restrict__ h0,     // [1,B,N*H]
                 const float* __restrict__ c0,     // [1,B,N*H]
                 const float* __restrict__ wih,    // [N,4H,I]
                 const float* __restrict__ whh,    // [N,4H,H]
                 const float* __restrict__ bih,    // [N,4H]
                 const float* __restrict__ bhh,    // [N,4H]
                 float* __restrict__ out)          // [B,T,N*H] ++ [B,N*H] ++ [B,N*H]
{
    extern __shared__ __align__(16) char smem[];
    unsigned short* wihsh  = (unsigned short*)(smem + OFF_WIH);
    unsigned short* whhsh  = (unsigned short*)(smem + OFF_WHH);
    float*          biassh = (float*)(smem + OFF_BIAS);
    unsigned short* xsh    = (unsigned short*)(smem + OFF_X);
    unsigned short* hsh    = (unsigned short*)(smem + OFF_HS);
    float*          csh    = (float*)(smem + OFF_CS);
    float*          gsh    = (float*)(smem + OFF_GATES);

    const int n    = blockIdx.x;          // LSTM block 0..31
    const int tid  = threadIdx.x;         // 0..511
    const int lane = tid & 31;
    const int wv   = tid >> 5;            // wave 0..15
    const int g    = lane >> 4;           // half-wave group
    const int l15  = lane & 15;

    float* out_hn = out + (size_t)LSTM_B * LSTM_T * (LSTM_N * LSTM_H);
    float* out_cn = out_hn + (size_t)LSTM_B * (LSTM_N * LSTM_H);

    // ---- one-time init: swizzle weights fp32->bf16 into WMMA B-fragment order
    // W_ih: idx = ((tile*2 + kc)*32 + lane)*16 + i ; B elem K = kc*32 + g*16 + i
    for (int idx = tid; idx < 32 * 2 * 512; idx += NTHREADS) {
        int i  = idx & 15;
        int ln = (idx >> 4) & 31;
        int kc = (idx >> 9) & 1;
        int tl = idx >> 10;
        int col = tl * 16 + (ln & 15);
        int k   = kc * 32 + (ln >> 4) * 16 + i;
        wihsh[idx] = f2bf(wih[(size_t)n * G4H * LSTM_I + (size_t)col * LSTM_I + k]);
    }
    // W_hh: idx = ((tile*4 + kc)*32 + lane)*16 + i
    for (int idx = tid; idx < 32 * 4 * 512; idx += NTHREADS) {
        int i  = idx & 15;
        int ln = (idx >> 4) & 31;
        int kc = (idx >> 9) & 3;
        int tl = idx >> 11;
        int col = tl * 16 + (ln & 15);
        int k   = kc * 32 + (ln >> 4) * 16 + i;
        whhsh[idx] = f2bf(whh[(size_t)n * G4H * LSTM_H + (size_t)col * LSTM_H + k]);
    }
    for (int idx = tid; idx < G4H; idx += NTHREADS)
        biassh[idx] = bih[n * G4H + idx] + bhh[n * G4H + idx];
    // h, c state
    for (int e = tid; e < LSTM_B * LSTM_H; e += NTHREADS) {
        int b = e >> 7, hh = e & 127;
        size_t gi = (size_t)b * (LSTM_N * LSTM_H) + n * LSTM_H + hh;
        csh[e] = c0[gi];
        hsh[e] = f2bf(h0[gi]);
    }
    // x_0
    for (int e = tid; e < LSTM_B * LSTM_I; e += NTHREADS) {
        int b = e >> 6, ii = e & 63;
        xsh[e] = f2bf(x[(size_t)(b * LSTM_T) * (LSTM_N * LSTM_I) + n * LSTM_I + ii]);
    }
    __syncthreads();

    const int t0 = wv * 2, t1 = wv * 2 + 1;   // this wave's N-tiles

    // ---- pin loop-invariant B (weight) fragments in VGPRs: 12 x v16bf/lane
    v16bf bfrag[12];
#pragma unroll
    for (int tix = 0; tix < 2; ++tix) {
        int tl = wv * 2 + tix;
#pragma unroll
        for (int kc = 0; kc < 6; ++kc) {
            const unsigned short* p =
                (kc < 2) ? &wihsh[((tl * 2 + kc) * 32 + lane) * 16]
                         : &whhsh[((tl * 4 + (kc - 2)) * 32 + lane) * 16];
            bfrag[tix * 6 + kc] = ld_frag32(p);
        }
    }
    const float bv0 = biassh[t0 * 16 + l15];
    const float bv1 = biassh[t1 * 16 + l15];

#pragma unroll 1
    for (int t = 0; t < LSTM_T; ++t) {
        // ---------------- GEMM phase: gates = [x_t | h] * [Wih | Whh]^T + bias
        v8f acc0 = {bv0, bv0, bv0, bv0, bv0, bv0, bv0, bv0};
        v8f acc1 = {bv1, bv1, bv1, bv1, bv1, bv1, bv1, bv1};

        // Issue ALL A-fragment LDS loads first; the sched_barrier keeps them
        // ahead of the WMMA chain so each value stays in its own VGPRs and the
        // WMMAs wait on partial dscnt thresholds while later loads complete.
        u16x8 araw[12];
#pragma unroll
        for (int kc = 0; kc < 6; ++kc) {
            const unsigned short* abase =
                (kc < 2) ? &xsh[l15 * LSTM_I + kc * 32]
                         : &hsh[l15 * LSTM_H + (kc - 2) * 32];
            araw[2 * kc]     = *(const u16x8*)(abase + g * 16);
            araw[2 * kc + 1] = *(const u16x8*)(abase + g * 16 + 8);
        }
        __builtin_amdgcn_sched_barrier(0);   // nothing crosses: loads stay hoisted

#pragma unroll
        for (int kc = 0; kc < 6; ++kc) {
            v16bf a = __builtin_bit_cast(
                v16bf, __builtin_shufflevector(araw[2 * kc], araw[2 * kc + 1],
                        0,1,2,3,4,5,6,7,8,9,10,11,12,13,14,15));
            acc0 = __builtin_amdgcn_wmma_f32_16x16x32_bf16(
                       false, a, false, bfrag[kc], (short)0, acc0, false, false);
            acc1 = __builtin_amdgcn_wmma_f32_16x16x32_bf16(
                       false, a, false, bfrag[6 + kc], (short)0, acc1, false, false);
        }
        // C/D layout: VGPR r -> row (g*8 + r), col = tile*16 + l15
#pragma unroll
        for (int r = 0; r < 8; ++r) {
            int row = g * 8 + r;
            gsh[row * G4H + t0 * 16 + l15] = acc0[r];
            gsh[row * G4H + t1 * 16 + l15] = acc1[r];
        }
        __syncthreads();

        // ---------------- elementwise phase (4 contiguous elems per thread)
        {
            const int e0  = tid * 4;                  // 0..2044
            const int b   = e0 >> 7;
            const int hh0 = e0 & 127;
            const float* gr = &gsh[b * G4H];
            v4f vi = *(const v4f*)&gr[hh0];
            v4f vf = *(const v4f*)&gr[LSTM_H + hh0];
            v4f vg = *(const v4f*)&gr[2 * LSTM_H + hh0];
            v4f vo = *(const v4f*)&gr[3 * LSTM_H + hh0];
            v4f vc = *(const v4f*)&csh[e0];
            v4f cn, hn;
            u16x4 hb;
#pragma unroll
            for (int q = 0; q < 4; ++q) {
                float ci = sigmoid_f(vf[q]) * vc[q] + sigmoid_f(vi[q]) * tanh_f(vg[q]);
                cn[q] = ci;
                float hv = sigmoid_f(vo[q]) * tanh_f(ci);
                hn[q] = hv;
                hb[q] = f2bf(hv);
            }
            *(v4f*)&csh[e0]   = cn;
            *(u16x4*)&hsh[e0] = hb;
            *(v4f*)&out[(size_t)(b * LSTM_T + t) * (LSTM_N * LSTM_H) + n * LSTM_H + hh0] = hn;
            if (t == LSTM_T - 1) {
                size_t oi = (size_t)b * (LSTM_N * LSTM_H) + n * LSTM_H + hh0;
                *(v4f*)&out_hn[oi] = hn;
                *(v4f*)&out_cn[oi] = cn;
            }
        }
        // ---------------- stage x_{t+1} (float2 per thread), prefetch x_{t+2}
        if (t + 1 < LSTM_T) {
            const int e0 = tid * 2;                   // 0..1022
            const int b  = e0 >> 6;
            const int ii = e0 & 63;
            const float* src =
                &x[(size_t)(b * LSTM_T + t + 1) * (LSTM_N * LSTM_I) + n * LSTM_I + ii];
            float2 xv = *(const float2*)src;
            unsigned int pk = (unsigned int)f2bf(xv.x) |
                              ((unsigned int)f2bf(xv.y) << 16);
            *(unsigned int*)&xsh[e0] = pk;
            if (t + 2 < LSTM_T)                       // global_prefetch_b8 of x_{t+2}
                __builtin_prefetch(src + (LSTM_N * LSTM_I), 0, 0);
        }
        __syncthreads();
    }
}

extern "C" void kernel_launch(void* const* d_in, const int* in_sizes, int n_in,
                              void* d_out, int out_size, void* d_ws, size_t ws_size,
                              hipStream_t stream) {
    (void)in_sizes; (void)n_in; (void)out_size; (void)d_ws; (void)ws_size;
    const float* x   = (const float*)d_in[0];
    const float* h0  = (const float*)d_in[1];
    const float* c0  = (const float*)d_in[2];
    const float* wih = (const float*)d_in[3];
    const float* whh = (const float*)d_in[4];
    const float* bih = (const float*)d_in[5];
    const float* bhh = (const float*)d_in[6];
    float* out = (float*)d_out;

    wide_lstm_kernel<<<dim3(LSTM_N), dim3(NTHREADS), SMEM_BYTES, stream>>>(
        x, h0, c0, wih, whh, bih, bhh, out);
}
